// NonCausalLayerLVC_12481174962929
// MI455X (gfx1250) — matrix operand order repaired and checked by reading, not compiled, not run on hardware
//
#include <hip/hip_runtime.h>
#include <hip/hip_bf16.h>

typedef __attribute__((ext_vector_type(16))) __bf16 v16bf;
typedef __attribute__((ext_vector_type(8)))  __bf16 v8bf;
typedef __attribute__((ext_vector_type(8)))  float  v8f;

#define TT   32768
#define SCH  128
#define OFFC 256
#define K1   192
#define LDA  200   // Aw row stride (bf16 elems)
#define LDXT 72    // xl_t row stride (bf16 elems), rows are time u=0..259, cols are channels c=0..63
#define LDW  72    // Wo row stride
#define LDG  72    // Gt row stride
#define LDSG 257   // sig row stride (floats)

// LDS layout (bytes). Wo/bias persistent; big region is a union of
// stage1 {Aw, xl_t} and stage2 {sig, Gt}.
#define OFF_WO   0
#define SZ_WO    (128*LDW*2)            // 18432
#define OFF_BIAS (OFF_WO + SZ_WO)
#define SZ_BIAS  (128*4)
#define OFF_U    (OFF_BIAS + SZ_BIAS)   // 18944
#define OFF_AW   (OFF_U)
#define SZ_AW    (128*LDA*2)            // 51200
#define OFF_XT   (OFF_AW + SZ_AW)       // 70144
#define SZ_XT    (260*LDXT*2)           // 37440 -> end 107584
#define OFF_SIG  (OFF_U)
#define SZ_SIG   (64*LDSG*4)            // 65792
#define OFF_GT   (OFF_SIG + SZ_SIG)     // 84736
#define SZ_GT    (256*LDG*2)            // 36864 -> end 121600
#define SMEM_BYTES 121600               // 2 WGs fit per 320KB WGP

__global__ __launch_bounds__(256) void lvc_wmma_kernel(
    const float* __restrict__ x,   const float* __restrict__ wt,
    const float* __restrict__ wo,  const float* __restrict__ wob,
    float* __restrict__ out)
{
  extern __shared__ char smem[];
  __bf16* sAw   = (__bf16*)(smem + OFF_AW);
  __bf16* sXt   = (__bf16*)(smem + OFF_XT);  // xl_t[u][c] = x[b,c,g0+u-2]
  __bf16* sWo   = (__bf16*)(smem + OFF_WO);
  float*  sBias = (float*) (smem + OFF_BIAS);
  float*  sSig  = (float*) (smem + OFF_SIG);
  __bf16* sGt   = (__bf16*)(smem + OFF_GT);

  const int tid = threadIdx.x;
  const int b   = blockIdx.x >> 7;
  const int s   = blockIdx.x & 127;
  const int g0  = s * OFFC;

  // ---- stage Wo, bias into LDS as bf16 ----
  for (int i = tid; i < 128*64; i += 256) {
    int o = i >> 6, c = i & 63;
    sWo[o*LDW + c] = (__bf16)wo[i];
  }
  if (tid < 128) sBias[tid] = wob[tid];

  // ---- stage Aw with K permuted r-major: Aw[o][r*64+c] = W[b,s,o,c,r] ----
  {
    const float* wsrc = wt + (((size_t)b*SCH + s) * 128) * K1;
    for (int i = tid; i < 128*K1; i += 256) {
      int o = i / K1, k = i - o*K1;      // k = 3c + r (memory order, coalesced)
      int c = k / 3,  r = k - 3*c;
      sAw[o*LDA + r*64 + c] = (__bf16)wsrc[i];
    }
  }

  // ---- stage transposed x window: xl_t[u][c], u in [0,260) ----
  {
    const float* xb = x + (size_t)b * 64 * TT;
    for (int i = tid; i < 64*260; i += 256) {
      int c = i / 260, u = i - c*260;    // consecutive i -> consecutive u (coalesced reads)
      int g = g0 + u - 2;
      float v = (g >= 0 && g < TT) ? xb[(size_t)c*TT + g] : 0.0f;
      sXt[u*LDXT + c] = (__bf16)v;
    }
  }
  __syncthreads();

  const int wv = tid >> 5;          // wave id 0..7
  const int mp = wv >> 1;           // M tile-pair 0..3 -> M tiles {2mp, 2mp+1}
  const int nh = wv & 1;            // N half: tiles [8*nh, 8*nh+8)
  const int lm = tid & 15;          // lane % 16
  const int lh = (tid >> 4) & 1;    // lane half

  union V16 { v16bf v; v8bf h[2]; };
  const v8f vzero = {0.f,0.f,0.f,0.f,0.f,0.f,0.f,0.f};

  // ---- GEMM1: Z(128x256) = Aw(128x192) x Taps(192x256), bf16 WMMA ----
  // K order is k' = r*64+c, so a 32-wide k-block has fixed r = ks>>1 and
  // B fragments are contiguous channel runs of xl_t at row u = t + 2r.
  v8f acc[16];
  #pragma unroll
  for (int i = 0; i < 16; ++i) acc[i] = vzero;
  {
    #pragma unroll
    for (int ks = 0; ks < 6; ++ks) {
      const int kA = ks*32 + lh*8;            // A: chunks {kA..+7} U {kA+16..+23}
      const int r  = ks >> 1;
      const int c0 = (ks & 1)*32 + lh*8;      // B: chunks {c0..+7} U {c0+16..+23}
      V16 a[2];
      #pragma unroll
      for (int mi = 0; mi < 2; ++mi) {
        const int mrow = mp*32 + mi*16 + lm;
        a[mi].h[0] = *(const v8bf*)(sAw + mrow*LDA + kA);
        a[mi].h[1] = *(const v8bf*)(sAw + mrow*LDA + kA + 16);
      }
      #pragma unroll
      for (int nt = 0; nt < 8; ++nt) {
        const int t = nh*128 + nt*16 + lm;    // B column = local time
        const int u = t + 2*r;
        V16 bm;
        bm.h[0] = *(const v8bf*)(sXt + u*LDXT + c0);
        bm.h[1] = *(const v8bf*)(sXt + u*LDXT + c0 + 16);
        #pragma unroll
        for (int mi = 0; mi < 2; ++mi)
          acc[mi*8+nt] = __builtin_amdgcn_wmma_f32_16x16x32_bf16(
              false, a[mi].v, false, bm.v, (short)0, acc[mi*8+nt], false, false);
      }
    }
  }
  __syncthreads();   // Aw/xl_t now dead; reuse region for sig/Gt

  // ---- gated activation: g = tanh(zw) * sigmoid(zv) (fast exp forms) ----
  if (mp >= 2) {     // these waves hold zv rows (Z rows 64..127)
    #pragma unroll
    for (int mi = 0; mi < 2; ++mi) {
      #pragma unroll
      for (int nt = 0; nt < 8; ++nt) {
        #pragma unroll
        for (int j = 0; j < 8; ++j) {
          const int op = (mp-2)*32 + mi*16 + j + 8*lh;
          const int t  = nh*128 + nt*16 + lm;
          const float z = acc[mi*8+nt][j];
          sSig[op*LDSG + t] = 1.0f / (1.0f + __expf(-z));
        }
      }
    }
  }
  __syncthreads();
  if (mp < 2) {      // these waves hold zw rows; write Gt[t][o] (B2^T layout)
    #pragma unroll
    for (int mi = 0; mi < 2; ++mi) {
      #pragma unroll
      for (int nt = 0; nt < 8; ++nt) {
        #pragma unroll
        for (int j = 0; j < 8; ++j) {
          const int o = mp*32 + mi*16 + j + 8*lh;
          const int t = nh*128 + nt*16 + lm;
          const float z  = acc[mi*8+nt][j];
          const float th = 2.0f / (1.0f + __expf(-2.0f*z)) - 1.0f;  // tanh(z)
          const float gg = th * sSig[o*LDSG + t];
          sGt[t*LDG + o] = (__bf16)gg;
        }
      }
    }
  }
  __syncthreads();

  // ---- GEMM2: Out(128x256) = Wo(128x64) x g(64x256) ----
  v8f acc2[16];
  #pragma unroll
  for (int i = 0; i < 16; ++i) acc2[i] = vzero;
  {
    #pragma unroll
    for (int ks = 0; ks < 2; ++ks) {
      const int kA = ks*32 + lh*8;
      const int kB = ks*32 + lh*8;
      V16 a[2];
      #pragma unroll
      for (int mi = 0; mi < 2; ++mi) {
        const int orow = mp*32 + mi*16 + lm;
        a[mi].h[0] = *(const v8bf*)(sWo + orow*LDW + kA);
        a[mi].h[1] = *(const v8bf*)(sWo + orow*LDW + kA + 16);
      }
      #pragma unroll
      for (int nt = 0; nt < 8; ++nt) {
        const int ncol = nh*128 + nt*16 + lm;
        V16 bm;
        bm.h[0] = *(const v8bf*)(sGt + ncol*LDG + kB);
        bm.h[1] = *(const v8bf*)(sGt + ncol*LDG + kB + 16);
        #pragma unroll
        for (int mi = 0; mi < 2; ++mi)
          acc2[mi*8+nt] = __builtin_amdgcn_wmma_f32_16x16x32_bf16(
              false, a[mi].v, false, bm.v, (short)0, acc2[mi*8+nt], false, false);
      }
    }
  }

  // ---- epilogue: + bias; rows<64 -> res (+x), rows>=64 -> skip ----
  const size_t skip_base = (size_t)4 * 64 * TT;
  #pragma unroll
  for (int mi = 0; mi < 2; ++mi) {
    #pragma unroll
    for (int j = 0; j < 8; ++j) {
      const int o = mp*32 + mi*16 + j + 8*lh;   // uniform <64 for mp<2
      const float bv = sBias[o];
      if (o < 64) {
        const float* xrow = x   + ((size_t)b*64 + o) * TT;
        float*       prow = out + ((size_t)b*64 + o) * TT;
        #pragma unroll
        for (int nt = 0; nt < 8; ++nt) {
          const int t = g0 + nh*128 + nt*16 + lm;
          prow[t] = acc2[mi*8+nt][j] + bv + xrow[t];
        }
      } else {
        float* prow = out + skip_base + ((size_t)b*64 + (o - 64)) * TT;
        #pragma unroll
        for (int nt = 0; nt < 8; ++nt) {
          const int t = g0 + nh*128 + nt*16 + lm;
          prow[t] = acc2[mi*8+nt][j] + bv;
        }
      }
    }
  }
}

extern "C" void kernel_launch(void* const* d_in, const int* in_sizes, int n_in,
                              void* d_out, int out_size, void* d_ws, size_t ws_size,
                              hipStream_t stream) {
  const float* x   = (const float*)d_in[0];
  const float* wt  = (const float*)d_in[1];
  const float* wo  = (const float*)d_in[2];
  const float* wob = (const float*)d_in[3];
  float* out = (float*)d_out;
  dim3 grid(4 * 128);   // one workgroup per (batch, chunk)
  lvc_wmma_kernel<<<grid, dim3(256), SMEM_BYTES, stream>>>(x, wt, wo, wob, out);
}
